// GINModel_64209761075678
// MI455X (gfx1250) — compile-verified, matrix operand order checked by reading, not run on hardware
//
#include <hip/hip_runtime.h>
#include <hip/hip_bf16.h>

#define N_NODES 100000
#define N_EDGES 1250000
#define FDIM 64
#define N_GRAPHS 64
#define WPAD 88   // padded LDS row stride in halves: 176B = 11*16B, aligned + low-conflict

typedef __attribute__((ext_vector_type(16))) _Float16 v16h;
typedef __attribute__((ext_vector_type(8)))  _Float16 v8h;
typedef __attribute__((ext_vector_type(8)))  float    v8f;

// ---------------------------------------------------------------------------
// agg = x  (vectorized copy, float4)
// ---------------------------------------------------------------------------
__global__ void gin_copy_kernel(const float* __restrict__ src,
                                float* __restrict__ dst, int n4) {
    int i = blockIdx.x * blockDim.x + threadIdx.x;
    if (i < n4) {
        reinterpret_cast<float4*>(dst)[i] =
            reinterpret_cast<const float4*>(src)[i];
    }
}

// ---------------------------------------------------------------------------
// agg[dst] += x[src]  : one wave32 per edge, 2 features per lane
// ---------------------------------------------------------------------------
__global__ void gin_scatter_kernel(const float* __restrict__ x,
                                   const long long* __restrict__ edge_index,
                                   float* __restrict__ agg) {
    int edge = (blockIdx.x * blockDim.x + threadIdx.x) >> 5;
    int lane = threadIdx.x & 31;
    if (edge >= N_EDGES) return;
    long long s = edge_index[edge];            // row 0: src
    long long d = edge_index[N_EDGES + edge];  // row 1: dst
    const float2 v = *reinterpret_cast<const float2*>(
        &x[(size_t)s * FDIM + lane * 2]);
    float* dp = &agg[(size_t)d * FDIM + lane * 2];
    atomicAdd(dp + 0, v.x);
    atomicAdd(dp + 1, v.y);
}

// ---------------------------------------------------------------------------
// out = relu(H @ W + b), H:[N_NODES,64] W:[64,64] b:[64]
// One wave per 16-row tile; 4 N-tiles x 2 K-fragments = 8 WMMA per wave.
// W staged TRANSPOSED in LDS as f16 so B fragments are contiguous 16B runs
// -> ds_load_b128 instead of scalar ds_load_u16 gathers.
// ---------------------------------------------------------------------------
__global__ __launch_bounds__(256)
void gin_gemm_relu_kernel(const float* __restrict__ H,
                          const float* __restrict__ W,
                          const float* __restrict__ bias,
                          float* __restrict__ out) {
    __shared__ __align__(16) _Float16 WsT[FDIM * WPAD]; // WsT[n*WPAD + k] = W[k*64+n]
    __shared__ float bs[FDIM];

    for (int i = threadIdx.x; i < FDIM * FDIM; i += blockDim.x) {
        int k = i >> 6;        // row of W (input feature)
        int n = i & 63;        // col of W (output feature)
        WsT[n * WPAD + k] = (_Float16)W[i];   // coalesced global read, scattered LDS write
    }
    if (threadIdx.x < FDIM) bs[threadIdx.x] = bias[threadIdx.x];
    __syncthreads();

    const int waveInBlock = threadIdx.x >> 5;
    const int lane        = threadIdx.x & 31;
    const int rowBase     = (blockIdx.x * 8 + waveInBlock) * 16;
    if (rowBase >= N_NODES) return;   // wave-uniform: EXEC all-ones at WMMA

    // ISA 7.12.2 16-bit A layout: lane = row (M); low half-wave holds
    // K = kb..kb+7 and kb+16..kb+23 with kb=0; high half-wave kb=8.
    const int m  = lane & 15;
    const int kb = (lane >> 4) * 8;
    const float* hrow = H + (size_t)(rowBase + m) * FDIM;

    v16h a0, a1;
#pragma unroll
    for (int i = 0; i < 8; ++i) {
        a0[i]     = (_Float16)hrow[kb + i];
        a0[8 + i] = (_Float16)hrow[kb + 16 + i];
        a1[i]     = (_Float16)hrow[32 + kb + i];
        a1[8 + i] = (_Float16)hrow[32 + kb + 16 + i];
    }

    const int hi = lane >> 4;
#pragma unroll
    for (int nt = 0; nt < 4; ++nt) {
        const int n = nt * 16 + m;                 // output column for this lane
        const _Float16* wrow = &WsT[n * WPAD];     // B layout mirrors A with lane = N
        v8h p0 = *reinterpret_cast<const v8h*>(wrow + kb);
        v8h p1 = *reinterpret_cast<const v8h*>(wrow + kb + 16);
        v8h p2 = *reinterpret_cast<const v8h*>(wrow + 32 + kb);
        v8h p3 = *reinterpret_cast<const v8h*>(wrow + 32 + kb + 16);
        v16h b0 = __builtin_shufflevector(p0, p1, 0,1,2,3,4,5,6,7,8,9,10,11,12,13,14,15);
        v16h b1 = __builtin_shufflevector(p2, p3, 0,1,2,3,4,5,6,7,8,9,10,11,12,13,14,15);

        v8f c = {};
        c = __builtin_amdgcn_wmma_f32_16x16x32_f16(false, a0, false, b0,
                                                   (short)0, c, false, false);
        c = __builtin_amdgcn_wmma_f32_16x16x32_f16(false, a1, false, b1,
                                                   (short)0, c, false, false);
        const float bv = bs[n];
#pragma unroll
        for (int j = 0; j < 8; ++j) {              // VGPR j: M = j + 8*hi
            float v = c[j] + bv;
            v = v > 0.0f ? v : 0.0f;
            out[(size_t)(rowBase + j + 8 * hi) * FDIM + n] = v;
        }
    }
}

// ---------------------------------------------------------------------------
// zero small accumulators
// ---------------------------------------------------------------------------
__global__ void gin_zero_kernel(float* __restrict__ p, int n) {
    int i = blockIdx.x * blockDim.x + threadIdx.x;
    if (i < n) p[i] = 0.0f;
}

// ---------------------------------------------------------------------------
// pooled[batch[node]] += h[node]; counts[batch[node]] += 1
// one wave per node, 2 features per lane
// ---------------------------------------------------------------------------
__global__ void gin_pool_kernel(const float* __restrict__ h,
                                const long long* __restrict__ batch,
                                float* __restrict__ pooled,
                                float* __restrict__ counts) {
    int node = (blockIdx.x * blockDim.x + threadIdx.x) >> 5;
    int lane = threadIdx.x & 31;
    if (node >= N_NODES) return;
    long long g = batch[node];
    const float2 v = *reinterpret_cast<const float2*>(
        &h[(size_t)node * FDIM + lane * 2]);
    float* dp = &pooled[(size_t)g * FDIM + lane * 2];
    atomicAdd(dp + 0, v.x);
    atomicAdd(dp + 1, v.y);
    if (lane == 0) atomicAdd(&counts[g], 1.0f);
}

// ---------------------------------------------------------------------------
// out[g] = (pooled[g]/max(counts[g],1)) . fcW + fcb
// ---------------------------------------------------------------------------
__global__ void gin_head_kernel(const float* __restrict__ pooled,
                                const float* __restrict__ counts,
                                const float* __restrict__ fcW,
                                const float* __restrict__ fcb,
                                float* __restrict__ out) {
    int g = threadIdx.x;
    if (g >= N_GRAPHS) return;
    float c = counts[g];
    c = c > 1.0f ? c : 1.0f;
    float acc = 0.0f;
    for (int f = 0; f < FDIM; ++f)
        acc += (pooled[g * FDIM + f] / c) * fcW[f];
    out[g] = acc + fcb[0];
}

// ---------------------------------------------------------------------------
extern "C" void kernel_launch(void* const* d_in, const int* in_sizes, int n_in,
                              void* d_out, int out_size, void* d_ws, size_t ws_size,
                              hipStream_t stream) {
    const float*     x     = (const float*)d_in[0];
    const long long* ei    = (const long long*)d_in[1];
    const long long* batch = (const long long*)d_in[2];
    const float* W1_0 = (const float*)d_in[3];
    const float* b1_0 = (const float*)d_in[4];
    const float* W2_0 = (const float*)d_in[5];
    const float* b2_0 = (const float*)d_in[6];
    const float* W1_1 = (const float*)d_in[7];
    const float* b1_1 = (const float*)d_in[8];
    const float* W2_1 = (const float*)d_in[9];
    const float* b2_1 = (const float*)d_in[10];
    const float* fcW  = (const float*)d_in[11];
    const float* fcb  = (const float*)d_in[12];
    float* out = (float*)d_out;

    const size_t nodeElems = (size_t)N_NODES * FDIM;       // 6.4M floats
    float* bufA   = (float*)d_ws;                          // agg / h2
    float* bufB   = bufA + nodeElems;                      // t1
    float* bufC   = bufB + nodeElems;                      // h1
    float* pooled = bufC + nodeElems;                      // 64*64
    float* counts = pooled + (size_t)N_GRAPHS * FDIM;      // 64

    const int threads = 256;
    const int copyBlocks    = (int)((nodeElems / 4 + threads - 1) / threads);
    const int scatterBlocks = (N_EDGES * 32 + threads - 1) / threads;
    const int gemmBlocks    = (N_NODES / 16 + 8 - 1) / 8;  // 8 waves/block
    const int poolBlocks    = (N_NODES * 32 + threads - 1) / threads;
    const int zeroN         = N_GRAPHS * FDIM + N_GRAPHS;
    const int zeroBlocks    = (zeroN + threads - 1) / threads;

    // ---- layer 0 ----
    gin_copy_kernel<<<copyBlocks, threads, 0, stream>>>(x, bufA, (int)(nodeElems / 4));
    gin_scatter_kernel<<<scatterBlocks, threads, 0, stream>>>(x, ei, bufA);
    gin_gemm_relu_kernel<<<gemmBlocks, threads, 0, stream>>>(bufA, W1_0, b1_0, bufB);
    gin_gemm_relu_kernel<<<gemmBlocks, threads, 0, stream>>>(bufB, W2_0, b2_0, bufC);

    // ---- layer 1 ----
    gin_copy_kernel<<<copyBlocks, threads, 0, stream>>>(bufC, bufA, (int)(nodeElems / 4));
    gin_scatter_kernel<<<scatterBlocks, threads, 0, stream>>>(bufC, ei, bufA);
    gin_gemm_relu_kernel<<<gemmBlocks, threads, 0, stream>>>(bufA, W1_1, b1_1, bufB);
    gin_gemm_relu_kernel<<<gemmBlocks, threads, 0, stream>>>(bufB, W2_1, b2_1, bufA);

    // ---- pooling + head ----
    gin_zero_kernel<<<zeroBlocks, threads, 0, stream>>>(pooled, zeroN);
    gin_pool_kernel<<<poolBlocks, threads, 0, stream>>>(bufA, batch, pooled, counts);
    gin_head_kernel<<<1, 64, 0, stream>>>(pooled, counts, fcW, fcb, out);
}
